// SelfAttentionBlock_20684562497742
// MI455X (gfx1250) — compile-verified, matrix operand order checked by reading, not run on hardware
//
#include <hip/hip_runtime.h>

// Problem constants (from reference)
#define Bn 4
#define Sn 2048
#define En 512
#define Hn 8
#define DHn 64
#define E3 1536

typedef __attribute__((ext_vector_type(16))) __bf16 v16bf;
typedef __attribute__((ext_vector_type(8)))  __bf16 bf16x8;
typedef __attribute__((ext_vector_type(8)))  float  v8f;

static __device__ inline v8f v8f_zero() {
  v8f z = {0.f, 0.f, 0.f, 0.f, 0.f, 0.f, 0.f, 0.f};
  return z;
}

static __device__ inline v8f wmma_bf16(v16bf a, v16bf b, v8f c) {
  // D = A(16x32 bf16) x B(32x16 bf16) + C(16x16 f32)
  return __builtin_amdgcn_wmma_f32_16x16x32_bf16(false, a, false, b, (short)0, c,
                                                 false, false);
}

// A/B operand fragment loader (16-bit 16x32 K-packing, wave32):
// lanes 0-15 hold K = {k0+0..7, k0+16..23}, lanes 16-31 hold K = {k0+8..15, k0+24..31}
// rowptr already points at this lane's row (M for A, N for B-transposed).
static __device__ inline v16bf load_frag(const __bf16* rowptr, int k0, int half) {
  const bf16x8* p0 = (const bf16x8*)(rowptr + k0 + half * 8);
  const bf16x8* p1 = (const bf16x8*)(rowptr + k0 + 16 + half * 8);
  bf16x8 lo = *p0;
  bf16x8 hi = *p1;
  v16bf r;
#pragma unroll
  for (int j = 0; j < 8; ++j) {
    r[j] = lo[j];
    r[j + 8] = hi[j];
  }
  return r;
}

// ---- gfx1250 async global->LDS data mover (ASYNCcnt path) ------------------
// 16B per lane: copies global memory into LDS; LDS byte address is the low 32 bits of
// the generic pointer (ISA 10.2: LDS aperture maps addr[31:0] to LDS offset).
static __device__ inline void async_copy_b128(__bf16* lds, const __bf16* gptr) {
  unsigned ldsoff = (unsigned)(uintptr_t)lds;
  asm volatile("global_load_async_to_lds_b128 %0, %1, off"
               :
               : "v"(ldsoff), "v"(gptr)
               : "memory");
}

static __device__ inline void wait_async0() {
#if defined(__has_builtin)
#if __has_builtin(__builtin_amdgcn_s_wait_asynccnt)
  __builtin_amdgcn_s_wait_asynccnt(0);
#else
  asm volatile("s_wait_asynccnt 0" ::: "memory");
#endif
#else
  asm volatile("s_wait_asynccnt 0" ::: "memory");
#endif
}

// Reductions across the 16 lanes of each wave half (wave32)
static __device__ inline float redmax16(float v) {
#pragma unroll
  for (int off = 8; off >= 1; off >>= 1) v = fmaxf(v, __shfl_xor(v, off, 32));
  return v;
}
static __device__ inline float redsum16(float v) {
#pragma unroll
  for (int off = 8; off >= 1; off >>= 1) v += __shfl_xor(v, off, 32);
  return v;
}

// ---------------------------------------------------------------------------
// Elementwise helpers
// ---------------------------------------------------------------------------
__global__ void f2bf_kernel(const float* __restrict__ in, __bf16* __restrict__ out, int n) {
  for (int i = blockIdx.x * blockDim.x + threadIdx.x; i < n;
       i += gridDim.x * blockDim.x)
    out[i] = (__bf16)in[i];
}

__global__ void zero_kernel(float* __restrict__ p, size_t n) {
  for (size_t i = blockIdx.x * (size_t)blockDim.x + threadIdx.x; i < n;
       i += (size_t)gridDim.x * blockDim.x)
    p[i] = 0.f;
}

// V transpose: qkv[b,s, 2E + h*64 + d] -> vt[((b*H+h)*64+d)*S + s]
__global__ void vtrans_kernel(const __bf16* __restrict__ qkv, __bf16* __restrict__ vt) {
  const int n = Bn * Sn * En;
  for (int idx = blockIdx.x * blockDim.x + threadIdx.x; idx < n;
       idx += gridDim.x * blockDim.x) {
    int row = idx >> 9;       // b*S + s
    int e = idx & 511;        // h*64 + d
    int b = row / Sn, s = row % Sn;
    int h = e >> 6, d = e & 63;
    vt[(size_t)((b * Hn + h) * DHn + d) * Sn + s] =
        qkv[(size_t)row * E3 + 2 * En + e];
  }
}

// ---------------------------------------------------------------------------
// LayerNorm: one block (256 thr) per row of 512; f32 in -> bf16 out
// ---------------------------------------------------------------------------
__global__ __launch_bounds__(256) void ln_kernel(const float* __restrict__ in,
                                                 const float* __restrict__ g,
                                                 const float* __restrict__ bt,
                                                 __bf16* __restrict__ out) {
  __shared__ float red[16];
  const int r = blockIdx.x;
  const int t = threadIdx.x;
  const float* xr = in + (size_t)r * En;
  float a0 = xr[t];
  float a1 = xr[t + 256];
  float s = a0 + a1;
  float ss = a0 * a0 + a1 * a1;
#pragma unroll
  for (int off = 16; off >= 1; off >>= 1) {
    s += __shfl_xor(s, off, 32);
    ss += __shfl_xor(ss, off, 32);
  }
  const int wid = t >> 5, lane = t & 31;
  if (lane == 0) {
    red[wid] = s;
    red[8 + wid] = ss;
  }
  __syncthreads();
  float S = 0.f, SS = 0.f;
#pragma unroll
  for (int i = 0; i < 8; ++i) {
    S += red[i];
    SS += red[8 + i];
  }
  const float mean = S * (1.f / (float)En);
  const float var = SS * (1.f / (float)En) - mean * mean;
  const float inv = rsqrtf(var + 1e-5f);
  out[(size_t)r * En + t] = (__bf16)((a0 - mean) * inv * g[t] + bt[t]);
  out[(size_t)r * En + t + 256] =
      (__bf16)((a1 - mean) * inv * g[t + 256] + bt[t + 256]);
}

// ---------------------------------------------------------------------------
// GEMM: C[M,N] = A[M,K](bf16) @ W[N,K](bf16)^T + bias, optional residual (f32),
// optional leaky-relu, f32 and/or bf16 output.
// Block: 256 thr = 8 waves as 2x4; wave = 32x32; block tile 64x128.
// A/B panels double-buffered in LDS via async global->LDS DMA (ASYNCcnt):
//   per k-step the block moves 12KB (A 64x32 + B 128x32) instead of each wave
//   re-reading 4KB from L2; waves compute stage s from LDS while the DMA
//   engine fills stage s+1.
// ---------------------------------------------------------------------------
__global__ __launch_bounds__(256) void gemm_kernel(
    const __bf16* __restrict__ A, const __bf16* __restrict__ W,
    const float* __restrict__ bias, const float* __restrict__ resid,
    float* __restrict__ outf, __bf16* __restrict__ outb, int M, int N, int K,
    int act) {
  __shared__ __bf16 sA[2][64 * 32];    // 8KB
  __shared__ __bf16 sB[2][128 * 32];   // 16KB
  const int tid = threadIdx.x;
  const int lane = tid & 31;
  const int wid = tid >> 5;
  const int wr = wid & 1, wc = wid >> 1;
  const int mb = blockIdx.x * 64;
  const int nb = blockIdx.y * 128;
  const int mw = mb + wr * 32;
  const int nw = nb + wc * 32;
  const int rl = lane & 15, half = lane >> 4;

  // Per-thread async-staging slices (16B chunks):
  //   B tile: 512 chunks -> c = tid, tid+256 ; A tile: 256 chunks -> c = tid
  const int bn0 = tid >> 2;                 // 0..63
  const int bn1 = (tid + 256) >> 2;         // 64..127
  const int bk = (tid & 3) * 8;             // 0,8,16,24
  const int am = tid >> 2;                  // 0..63
  const int ak = bk;

  v8f acc00 = v8f_zero(), acc01 = v8f_zero();
  v8f acc10 = v8f_zero(), acc11 = v8f_zero();

  const int nsteps = K / 32;

  // prologue: stage 0
  {
    const int k0 = 0;
    async_copy_b128(&sB[0][bn0 * 32 + bk], W + (size_t)(nb + bn0) * K + k0 + bk);
    async_copy_b128(&sB[0][bn1 * 32 + bk], W + (size_t)(nb + bn1) * K + k0 + bk);
    async_copy_b128(&sA[0][am * 32 + ak], A + (size_t)(mb + am) * K + k0 + ak);
  }

  for (int s = 0; s < nsteps; ++s) {
    wait_async0();      // this wave's staged chunks have landed in LDS
    __syncthreads();    // everyone's chunks landed; previous stage fully read
    if (s + 1 < nsteps) {
      const int k0 = (s + 1) * 32;
      const int nxt = (s + 1) & 1;
      async_copy_b128(&sB[nxt][bn0 * 32 + bk],
                      W + (size_t)(nb + bn0) * K + k0 + bk);
      async_copy_b128(&sB[nxt][bn1 * 32 + bk],
                      W + (size_t)(nb + bn1) * K + k0 + bk);
      async_copy_b128(&sA[nxt][am * 32 + ak],
                      A + (size_t)(mb + am) * K + k0 + ak);
    }
    const int cur = s & 1;
    const __bf16* pa = &sA[cur][(wr * 32 + rl) * 32];
    const __bf16* pb = &sB[cur][(wc * 32 + rl) * 32];
    v16bf a0 = load_frag(pa, 0, half);
    v16bf a1 = load_frag(pa + 16 * 32, 0, half);
    v16bf b0 = load_frag(pb, 0, half);
    v16bf b1 = load_frag(pb + 16 * 32, 0, half);
    acc00 = wmma_bf16(a0, b0, acc00);
    acc01 = wmma_bf16(a0, b1, acc01);
    acc10 = wmma_bf16(a1, b0, acc10);
    acc11 = wmma_bf16(a1, b1, acc11);
  }

  v8f* accs[4] = {&acc00, &acc01, &acc10, &acc11};
#pragma unroll
  for (int t = 0; t < 4; ++t) {
    const int ti = t >> 1, tj = t & 1;
#pragma unroll
    for (int i = 0; i < 8; ++i) {
      const int m = mw + ti * 16 + half * 8 + i;  // D-layout: M = vgpr + 8*half
      const int n = nw + tj * 16 + rl;            // N = lane%16
      float v = (*accs[t])[i] + bias[n];
      if (resid) v += resid[(size_t)m * N + n];
      if (act) v = v > 0.f ? v : 0.01f * v;
      if (outf) outf[(size_t)m * N + n] = v;
      if (outb) outb[(size_t)m * N + n] = (__bf16)v;
    }
  }
}

// ---------------------------------------------------------------------------
// Causal attention: one wave per (b, 16-query tile), loops over all 8 heads.
// Two-pass softmax (online max/sum, then prob + P@V). Probabilities routed
// through a per-wave LDS tile to convert D-layout -> A-operand layout, and
// accumulated (/H) into the attention-map output (deterministic, no atomics:
// each row block is owned by exactly one wave).
// ---------------------------------------------------------------------------
__global__ __launch_bounds__(256) void attn_kernel(const __bf16* __restrict__ qkv,
                                                   const __bf16* __restrict__ vt,
                                                   __bf16* __restrict__ ctx,
                                                   float* __restrict__ attnw) {
  __shared__ __bf16 pl[8][16 * 32];  // 8 waves x (16 q x 32 k) bf16 = 8KB
  const int tid = threadIdx.x;
  const int lane = tid & 31;
  const int wslot = tid >> 5;
  const int gw = blockIdx.x * 8 + wslot;  // 0 .. B*S/16-1
  const int b = gw / (Sn / 16);
  const int qt = gw % (Sn / 16);
  const int qb = qt * 16;
  const int rl = lane & 15;
  const int half = lane >> 4;
  __bf16* plds = pl[wslot];
  float* aw = attnw + (size_t)b * Sn * Sn;

  for (int h = 0; h < Hn; ++h) {
    const __bf16* qbase = qkv + (size_t)(b * Sn + qb + rl) * E3 + h * DHn;
    v16bf qf0 = load_frag(qbase, 0, half);
    v16bf qf1 = load_frag(qbase, 32, half);

    float rmax[8], rsum[8];
#pragma unroll
    for (int i = 0; i < 8; ++i) {
      rmax[i] = -3.0e38f;
      rsum[i] = 0.f;
    }

    // ---- pass 1: online row-max / row-sum over all unmasked key tiles ----
    for (int kb = 0; kb <= qb; kb += 16) {
      const __bf16* kbase =
          qkv + (size_t)(b * Sn + kb + rl) * E3 + En + h * DHn;
      __builtin_prefetch(kbase + 16 * E3, 0, 3);  // next K tile row
      v16bf kf0 = load_frag(kbase, 0, half);
      v16bf kf1 = load_frag(kbase, 32, half);
      v8f acc = v8f_zero();
      acc = wmma_bf16(qf0, kf0, acc);
      acc = wmma_bf16(qf1, kf1, acc);
      const bool diag = (kb == qb);
#pragma unroll
      for (int i = 0; i < 8; ++i) {
        const int m = i + half * 8;
        float s = acc[i] * 0.125f;  // 1/sqrt(64)
        if (diag && rl > m) s = -1.0e30f;
        float tmax = redmax16(s);
        float mnew = fmaxf(rmax[i], tmax);
        float tsum = redsum16(__expf(s - mnew));
        rsum[i] = rsum[i] * __expf(rmax[i] - mnew) + tsum;
        rmax[i] = mnew;
      }
    }

    // ---- pass 2: probabilities -> attn map output and P @ V ----
    v8f o[4];
#pragma unroll
    for (int c = 0; c < 4; ++c) o[c] = v8f_zero();

    for (int ks = 0; ks <= qb; ks += 32) {
#pragma unroll
      for (int sub = 0; sub < 2; ++sub) {
        const int kb = ks + sub * 16;
        if (kb <= qb) {
          const __bf16* kbase =
              qkv + (size_t)(b * Sn + kb + rl) * E3 + En + h * DHn;
          v16bf kf0 = load_frag(kbase, 0, half);
          v16bf kf1 = load_frag(kbase, 32, half);
          v8f acc = v8f_zero();
          acc = wmma_bf16(qf0, kf0, acc);
          acc = wmma_bf16(qf1, kf1, acc);
          const bool diag = (kb == qb);
#pragma unroll
          for (int i = 0; i < 8; ++i) {
            const int m = i + half * 8;
            float s = acc[i] * 0.125f;
            if (diag && rl > m) s = -1.0e30f;
            float p = __expf(s - rmax[i]) / rsum[i];
            plds[m * 32 + sub * 16 + rl] = (__bf16)p;
            aw[(size_t)(qb + m) * Sn + kb + rl] += p * (1.f / (float)Hn);
          }
        } else {  // fully masked sub-tile: zero its LDS half
#pragma unroll
          for (int i = 0; i < 8; ++i) {
            const int m = i + half * 8;
            plds[m * 32 + sub * 16 + rl] = (__bf16)0.f;
          }
        }
      }
      // P tile (16q x 32k) as A operand from LDS
      v16bf pf = load_frag(plds + rl * 32, 0, half);
#pragma unroll
      for (int c = 0; c < 4; ++c) {
        const __bf16* vbase =
            vt + (size_t)((b * Hn + h) * DHn + c * 16 + rl) * Sn + ks;
        v16bf vf = load_frag(vbase, 0, half);
        o[c] = wmma_bf16(pf, vf, o[c]);
      }
    }

    // store ctx (bf16) for the out-projection GEMM
#pragma unroll
    for (int c = 0; c < 4; ++c)
#pragma unroll
      for (int i = 0; i < 8; ++i) {
        const int m = i + half * 8;
        ctx[(size_t)(b * Sn + qb + m) * En + h * DHn + c * 16 + rl] =
            (__bf16)o[c][i];
      }
  }
}

// ---------------------------------------------------------------------------
// Launch
// ---------------------------------------------------------------------------
extern "C" void kernel_launch(void* const* d_in, const int* in_sizes, int n_in,
                              void* d_out, int out_size, void* d_ws,
                              size_t ws_size, hipStream_t stream) {
  (void)in_sizes; (void)n_in; (void)out_size; (void)ws_size;
  const float* x     = (const float*)d_in[0];
  const float* ln1_g = (const float*)d_in[1];
  const float* ln1_b = (const float*)d_in[2];
  const float* w_qkv = (const float*)d_in[3];
  const float* b_qkv = (const float*)d_in[4];
  const float* w_out = (const float*)d_in[5];
  const float* b_out = (const float*)d_in[6];
  const float* ln2_g = (const float*)d_in[7];
  const float* ln2_b = (const float*)d_in[8];
  const float* w_fc1 = (const float*)d_in[9];
  const float* b_fc1 = (const float*)d_in[10];
  const float* w_fc2 = (const float*)d_in[11];
  const float* b_fc2 = (const float*)d_in[12];

  float* y_out = (float*)d_out;
  float* aw_out = y_out + (size_t)Bn * Sn * En;

  char* ws = (char*)d_ws;
  size_t off = 0;
  auto carve = [&](size_t bytes) -> char* {
    char* p = ws + off;
    off += (bytes + 255) & ~(size_t)255;
    return p;
  };
  __bf16* zbf   = (__bf16*)carve((size_t)Bn * Sn * En * 2);   // LN1 out
  __bf16* wqkvb = (__bf16*)carve((size_t)E3 * En * 2);
  __bf16* woutb = (__bf16*)carve((size_t)En * En * 2);
  __bf16* wfc1b = (__bf16*)carve((size_t)En * En * 2);
  __bf16* wfc2b = (__bf16*)carve((size_t)En * En * 2);
  __bf16* qkvb  = (__bf16*)carve((size_t)Bn * Sn * E3 * 2);   // QKV
  __bf16* vtb   = (__bf16*)carve((size_t)Bn * Sn * En * 2);   // V^T [b,h,d,s]
  __bf16* ctxb  = (__bf16*)carve((size_t)Bn * Sn * En * 2);   // attention ctx
  float*  z2    = (float*)carve((size_t)Bn * Sn * En * 4);    // residual 1
  __bf16* h0b   = (__bf16*)carve((size_t)Bn * Sn * En * 2);   // LN2 out
  __bf16* h1b   = (__bf16*)carve((size_t)Bn * Sn * En * 2);   // FC1 out

  const int T = 256;
  const int Mrows = Bn * Sn;  // 8192

  // 1) weight f32 -> bf16
  f2bf_kernel<<<1024, T, 0, stream>>>(w_qkv, wqkvb, E3 * En);
  f2bf_kernel<<<512, T, 0, stream>>>(w_out, woutb, En * En);
  f2bf_kernel<<<512, T, 0, stream>>>(w_fc1, wfc1b, En * En);
  f2bf_kernel<<<512, T, 0, stream>>>(w_fc2, wfc2b, En * En);
  // 2) LN1
  ln_kernel<<<Mrows, T, 0, stream>>>(x, ln1_g, ln1_b, zbf);
  // 3) QKV GEMM  (8192 x 1536 x 512)
  gemm_kernel<<<dim3(Mrows / 64, E3 / 128), T, 0, stream>>>(
      zbf, wqkvb, b_qkv, nullptr, nullptr, qkvb, Mrows, E3, En, 0);
  // 4) V transpose for the P@V B-operand
  vtrans_kernel<<<2048, T, 0, stream>>>(qkvb, vtb);
  // 5) zero the attention-map output (poisoned by harness; we accumulate +=)
  zero_kernel<<<4096, T, 0, stream>>>(aw_out, (size_t)Bn * Sn * Sn);
  // 6) attention (512 waves, 8 per block)
  attn_kernel<<<(Bn * (Sn / 16)) / 8, T, 0, stream>>>(qkvb, vtb, ctxb, aw_out);
  // 7) out-projection + residual(x) -> z2 (f32)
  gemm_kernel<<<dim3(Mrows / 64, En / 128), T, 0, stream>>>(
      ctxb, woutb, b_out, x, z2, nullptr, Mrows, En, En, 0);
  // 8) LN2
  ln_kernel<<<Mrows, T, 0, stream>>>(z2, ln2_g, ln2_b, h0b);
  // 9) FC1 + leaky-relu -> bf16
  gemm_kernel<<<dim3(Mrows / 64, En / 128), T, 0, stream>>>(
      h0b, wfc1b, b_fc1, nullptr, nullptr, h1b, Mrows, En, En, 1);
  // 10) FC2 + residual(z2) -> y (f32, straight into d_out)
  gemm_kernel<<<dim3(Mrows / 64, En / 128), T, 0, stream>>>(
      h1b, wfc2b, b_fc2, z2, y_out, nullptr, Mrows, En, En, 0);
}
